// BootstrappedCE_33663953666554
// MI455X (gfx1250) — compile-verified, matrix operand order checked by reading, not run on hardware
//
#include <hip/hip_runtime.h>
#include <math.h>

typedef __attribute__((ext_vector_type(16))) _Float16 v16h;
typedef __attribute__((ext_vector_type(8)))  float    v8f;

// Problem constants (trace-time constants in the reference)
constexpr int CB   = 8;            // batch
constexpr int CC   = 8;            // classes
constexpr int CHW_ = 512 * 512;    // H*W = 262144 = 2^18
constexpr int NPIX = CB * CHW_;    // 2097152 = 2^21
constexpr int KSEL = 314572;       // int(0.15 * 2097152)
constexpr int NBIN = 2048;         // 11-bit radix buckets
constexpr int NWG_SUM = 2048;

// ---------------------------------------------------------------------------
__global__ __launch_bounds__(256) void zero_kernel(unsigned* __restrict__ ghist) {
    int j = blockIdx.x * 256 + threadIdx.x;
    if (j < CC * NBIN) ghist[j] = 0u;
}

// ---------------------------------------------------------------------------
__device__ __forceinline__ unsigned ce_key(const float x[CC], int t) {
    float m = -INFINITY;
    #pragma unroll
    for (int c = 0; c < CC; ++c) m = fmaxf(m, x[c]);
    float ssum = 0.0f;
    #pragma unroll
    for (int c = 0; c < CC; ++c) ssum += __expf(x[c] - m);
    const float lse = m + __logf(ssum);
    float ce = lse - x[t];
    ce = fmaxf(ce, 0.0f);                 // CE >= 0 so uint order == float order
    return __float_as_uint(ce);
}

// Fused: per-pixel CE (log-softmax gather), key/class store, level-0 histogram.
// 4 consecutive pixels per thread-iteration -> b128 loads/stores everywhere.
__global__ __launch_bounds__(256) void ce_hist_kernel(const float* __restrict__ logits,
                                                      const int* __restrict__ tgt,
                                                      unsigned* __restrict__ keys,
                                                      unsigned* __restrict__ clsw,
                                                      unsigned* __restrict__ ghist) {
    __shared__ unsigned hist[CC * NBIN];   // 64 KiB LDS
    const int tid = threadIdx.x;
    for (int j = tid; j < CC * NBIN; j += 256) hist[j] = 0u;
    __syncthreads();

    const int g  = blockIdx.x * 256 + tid;   // 1024 WGs * 256 = 262144 threads
    const int TT = 262144;
    #pragma unroll
    for (int s = 0; s < 2; ++s) {            // 2 groups of 4 pixels per thread
        const int q  = s * TT + g;           // group index, coalesced
        const int p  = q * 4;                // base pixel; 4 pixels share batch b
        const int b  = p >> 18;
        const int hw = p & (CHW_ - 1);
        const float* base = logits + ((size_t)b << 21) + hw;

        float x0[CC], x1[CC], x2[CC], x3[CC];
        #pragma unroll
        for (int c = 0; c < CC; ++c) {
            const float4 v = *(const float4*)(base + ((size_t)c << 18));  // global_load_b128
            x0[c] = v.x; x1[c] = v.y; x2[c] = v.z; x3[c] = v.w;
        }
        const int4 t4 = *(const int4*)(tgt + p);                          // global_load_b128
        const int t0 = t4.x & 7, t1 = t4.y & 7, t2 = t4.z & 7, t3 = t4.w & 7;

        uint4 kv;
        kv.x = ce_key(x0, t0);
        kv.y = ce_key(x1, t1);
        kv.z = ce_key(x2, t2);
        kv.w = ce_key(x3, t3);
        *(uint4*)(keys + p) = kv;                                         // global_store_b128
        clsw[q] = (unsigned)t0 | ((unsigned)t1 << 8) |
                  ((unsigned)t2 << 16) | ((unsigned)t3 << 24);            // packed uchar4

        atomicAdd(&hist[t0 * NBIN + (kv.x >> 21)], 1u);
        atomicAdd(&hist[t1 * NBIN + (kv.y >> 21)], 1u);
        atomicAdd(&hist[t2 * NBIN + (kv.z >> 21)], 1u);
        atomicAdd(&hist[t3 * NBIN + (kv.w >> 21)], 1u);
    }
    __syncthreads();
    for (int j = tid; j < CC * NBIN; j += 256) {
        const unsigned v = hist[j];
        if (v) atomicAdd(&ghist[j], v);
    }
}

// ---------------------------------------------------------------------------
// Conditional histogram passes (levels 1 and 2 of the radix select)
template <int LEVEL>
__global__ __launch_bounds__(256) void hist_kernel(const unsigned* __restrict__ keys,
                                                   const unsigned* __restrict__ clsw,
                                                   unsigned* __restrict__ ghist,
                                                   const unsigned* __restrict__ state) {
    __shared__ unsigned hist[CC * NBIN];
    __shared__ unsigned pfx[CC];
    const int tid = threadIdx.x;
    for (int j = tid; j < CC * NBIN; j += 256) hist[j] = 0u;
    if (tid < CC) pfx[tid] = state[tid];
    __syncthreads();

    const int g  = blockIdx.x * 256 + tid;
    const int TT = 262144;
    #pragma unroll
    for (int s = 0; s < 2; ++s) {
        const int q = s * TT + g;
        const uint4 kv = *(const uint4*)(keys + q * 4);   // global_load_b128
        const unsigned cw = clsw[q];
        const unsigned kk[4] = {kv.x, kv.y, kv.z, kv.w};
        #pragma unroll
        for (int i = 0; i < 4; ++i) {
            const unsigned key = kk[i];
            const int c = (int)((cw >> (8 * i)) & 7u);
            if (LEVEL == 1) {
                if ((key >> 21) == pfx[c])
                    atomicAdd(&hist[c * NBIN + ((key >> 10) & 0x7FFu)], 1u);
            } else {
                if ((key >> 10) == pfx[c])
                    atomicAdd(&hist[c * NBIN + (key & 0x3FFu)], 1u);
            }
        }
    }
    __syncthreads();
    for (int j = tid; j < CC * NBIN; j += 256) {
        const unsigned v = hist[j];
        if (v) atomicAdd(&ghist[j], v);
    }
}

// ---------------------------------------------------------------------------
// Single-WG scan: pick the bucket containing rank `rem` (descending), update
// per-class prefix/remaining; implicit zeros (other classes' row positions)
// live in bucket 0 whenever the class prefix is all-zero. Then re-zero ghist.
template <int LEVEL>
__global__ __launch_bounds__(256) void scan_kernel(unsigned* __restrict__ ghist,
                                                   unsigned* __restrict__ state) {
    const int tid = threadIdx.x;
    if (tid < CC) {
        const int c = tid;
        unsigned pfx, rem, zeros;
        if (LEVEL == 0) {
            unsigned n = 0;
            for (int b = 0; b < NBIN; ++b) n += ghist[c * NBIN + b];
            zeros = (unsigned)NPIX - n;   // implicit zeros in this class's row
            rem   = (unsigned)KSEL;
            pfx   = 0u;
        } else {
            pfx = state[c]; rem = state[8 + c]; zeros = state[16 + c];
        }
        const int nb = (LEVEL == 2) ? 1024 : NBIN;
        unsigned cum = 0, sel = 0;
        for (int b = nb - 1; b >= 0; --b) {
            unsigned cnt = ghist[c * NBIN + b];
            if (b == 0 && pfx == 0u) cnt += zeros;
            if (cum + cnt >= rem) { sel = (unsigned)b; break; }
            cum += cnt;
        }
        rem -= cum;   // rank within the selected bucket (>= 1)
        const unsigned np = (LEVEL == 0) ? sel
                          : (LEVEL == 1) ? ((pfx << 11) | sel)
                                         : ((pfx << 10) | sel);  // full 32-bit key T_c
        state[c]     = np;
        state[8 + c] = rem;
        if (LEVEL == 0) state[16 + c] = zeros;
    }
    __syncthreads();
    for (int j = tid; j < CC * NBIN; j += 256) ghist[j] = 0u;
}

// ---------------------------------------------------------------------------
// Thresholded per-class sums. The 256-thread x 8-class workgroup reduction is
// done on the matrix pipe: A rows 0..7 carry class partials, B = ones, so
// D[c][0] = workgroup sum for class c (v_wmma_f32_16x16x32_f16).
__global__ __launch_bounds__(256) void sum_kernel(const unsigned* __restrict__ keys,
                                                  const unsigned* __restrict__ clsw,
                                                  const unsigned* __restrict__ state,
                                                  float* __restrict__ partials) {
    __shared__ float sdata[256 * CC];
    __shared__ unsigned T[CC];
    const int tid = threadIdx.x;
    if (tid < CC) T[tid] = state[tid];
    __syncthreads();

    float acc[CC];
    #pragma unroll
    for (int c = 0; c < CC; ++c) acc[c] = 0.0f;

    const int g = blockIdx.x * 256 + tid;    // 2048 WGs * 256 = 524288 threads, 4 px each
    {
        const uint4 kv = *(const uint4*)(keys + g * 4);   // global_load_b128
        const unsigned cw = clsw[g];
        const unsigned kk[4] = {kv.x, kv.y, kv.z, kv.w};
        #pragma unroll
        for (int i = 0; i < 4; ++i) {
            const unsigned key = kk[i];
            const int c = (int)((cw >> (8 * i)) & 7u);
            if (key > T[c]) acc[c] += __uint_as_float(key);  // strictly above threshold
        }
    }
    #pragma unroll
    for (int c = 0; c < CC; ++c) sdata[tid * CC + c] = acc[c];
    __syncthreads();

    if (tid < 32) {                 // wave 0: uniform condition -> EXEC all ones at WMMA
        const int m  = tid & 15;
        const int hi = tid >> 4;
        v16h a, ones;
        v8f  cacc;
        #pragma unroll
        for (int h = 0; h < 16; ++h) {
            float sv = 0.0f;
            if (m < CC) {
                const int grp = h + hi * 16;           // 32 groups of 8 threads
                #pragma unroll
                for (int j = 0; j < 8; ++j) sv += sdata[(grp * 8 + j) * CC + m];
            }
            a[h]    = (_Float16)sv;
            ones[h] = (_Float16)1.0f;
        }
        #pragma unroll
        for (int r = 0; r < 8; ++r) cacc[r] = 0.0f;
        cacc = __builtin_amdgcn_wmma_f32_16x16x32_f16(
            /*neg_a=*/false, a, /*neg_b=*/false, ones,
            /*c_mod=*/(short)0, cacc, /*reuse_a=*/false, /*reuse_b=*/false);
        // D layout: VGPR r, lanes 0-15 hold M=r, N=lane => lane 0 has D[r][0] = class-r sum
        if (tid == 0) {
            #pragma unroll
            for (int r = 0; r < CC; ++r) partials[(size_t)blockIdx.x * CC + r] = cacc[r];
        }
    }
}

// ---------------------------------------------------------------------------
// Deterministic fixed-order final reduction: mean = sum_c (S_c + rem_c*T_c) / (C*k)
__global__ __launch_bounds__(64) void final_kernel(const float* __restrict__ partials,
                                                   const unsigned* __restrict__ state,
                                                   float* __restrict__ out) {
    __shared__ float csum[CC];
    const int tid = threadIdx.x;
    if (tid < CC) {
        float s = 0.0f;
        for (int w = 0; w < NWG_SUM; ++w) s += partials[(size_t)w * CC + tid];
        const float Tf  = __uint_as_float(state[tid]);
        const float rem = (float)state[8 + tid];
        csum[tid] = s + rem * Tf;    // ties at T contribute rem copies of T exactly
    }
    __syncthreads();
    if (tid == 0) {
        float r = 0.0f;
        #pragma unroll
        for (int c = 0; c < CC; ++c) r += csum[c];
        out[0] = r / (8.0f * 314572.0f);
    }
}

// ---------------------------------------------------------------------------
extern "C" void kernel_launch(void* const* d_in, const int* in_sizes, int n_in,
                              void* d_out, int out_size, void* d_ws, size_t ws_size,
                              hipStream_t stream) {
    (void)in_sizes; (void)n_in; (void)out_size; (void)ws_size;
    const float* logits = (const float*)d_in[0];
    const int*   tgt    = (const int*)d_in[1];

    char* ws = (char*)d_ws;
    unsigned* keys     = (unsigned*)ws;                                       // 8 MiB
    unsigned* clsw     = (unsigned*)(ws + (size_t)NPIX * 4);                  // 2 MiB (uchar4 packed)
    unsigned* ghist    = (unsigned*)(ws + (size_t)NPIX * 5);                  // 64 KiB
    unsigned* state    = (unsigned*)(ws + (size_t)NPIX * 5 + CC * NBIN * 4);  // 128 B
    float*    partials = (float*)(ws + (size_t)NPIX * 5 + CC * NBIN * 4 + 128); // 64 KiB

    zero_kernel<<<64, 256, 0, stream>>>(ghist);
    ce_hist_kernel<<<1024, 256, 0, stream>>>(logits, tgt, keys, clsw, ghist);
    scan_kernel<0><<<1, 256, 0, stream>>>(ghist, state);
    hist_kernel<1><<<1024, 256, 0, stream>>>(keys, clsw, ghist, state);
    scan_kernel<1><<<1, 256, 0, stream>>>(ghist, state);
    hist_kernel<2><<<1024, 256, 0, stream>>>(keys, clsw, ghist, state);
    scan_kernel<2><<<1, 256, 0, stream>>>(ghist, state);
    sum_kernel<<<NWG_SUM, 256, 0, stream>>>(keys, clsw, state, partials);
    final_kernel<<<1, 64, 0, stream>>>(partials, state, (float*)d_out);
}
